// SDPAttention_73108933312945
// MI455X (gfx1250) — matrix operand order repaired and checked by reading, not compiled, not run on hardware
//
#include <hip/hip_runtime.h>
#include <stdint.h>

// ---------------------------------------------------------------------------
// CDNA5 (gfx1250) fused attention: QKV proj -> RoPE -> flash attention -> out
// proj.  GEMMs via v_wmma_f32_16x16x32_bf16 (wave32).  bf16 activation path.
// Data movement: per-lane async DMA (global_load_async_to_lds_b128, ASYNCcnt)
// for Q/GEMM-A tiles, Tensor Data Mover (tensor_load_to_lds, TENSORcnt) for
// flash-attention K tiles with hardware LDS row padding.
// ---------------------------------------------------------------------------

typedef __attribute__((ext_vector_type(16))) __bf16   bf16x16;
typedef __attribute__((ext_vector_type(8)))  float    floatx8;
typedef __attribute__((ext_vector_type(4)))  unsigned uint4s;
typedef __attribute__((ext_vector_type(8)))  unsigned uint8s;

union Frag {
    bf16x16  v;
    unsigned u[8];
};

__device__ __forceinline__ unsigned short f2bf(float f) {
    return __builtin_bit_cast(unsigned short, static_cast<__bf16>(f));
}
__device__ __forceinline__ float bf2f(unsigned short u) {
    return static_cast<float>(__builtin_bit_cast(__bf16, u));
}

__device__ __forceinline__ floatx8 wmma_bf16(const Frag& a, const Frag& b, floatx8 c) {
    return __builtin_amdgcn_wmma_f32_16x16x32_bf16(
        false, a.v, false, b.v, (short)0, c, false, false);
}

// low 32 bits of a flat pointer into LDS == LDS byte address (aperture rule)
__device__ __forceinline__ unsigned lds_off32(const void* p) {
    return (unsigned)(uintptr_t)p;
}
// async DMA: 16 bytes global -> LDS, tracked by ASYNCcnt
__device__ __forceinline__ void async_copy_b128(const void* g, unsigned lds) {
    asm volatile("global_load_async_to_lds_b128 %0, %1, off"
                 :: "v"(lds), "v"((unsigned long long)(uintptr_t)g)
                 : "memory");
}
__device__ __forceinline__ void wait_asynccnt0() {
    asm volatile("s_wait_asynccnt 0x0" ::: "memory");
}

// ---------------------------------------------------------------------------
// Tensor Data Mover: DMA a 2D tile (bf16) global -> LDS, one instruction per
// wave.  D# built per ISA 8.3/8.4: group0 = {count, lds_addr, global_addr,
// type=2}, group1 = {data_size=2B, pad config, tensor/tile dims, row stride}.
// pad_interval_code / pad_amount_code implement the padded LDS row layout.
// ---------------------------------------------------------------------------
__device__ __forceinline__ void tdm_load_2d_bf16(
    unsigned lds_byte_addr, const void* gaddr,
    unsigned tile_d0, unsigned tile_d1, unsigned row_stride_elems,
    unsigned pad_interval_code, unsigned pad_amount_code)
{
    unsigned long long ga = (unsigned long long)(uintptr_t)gaddr;
    uint4s g0;
    g0.x = 1u;                                            // count=1 (valid), user desc
    g0.y = lds_byte_addr;                                 // LDS dest (bytes)
    g0.z = (unsigned)(ga & 0xffffffffu);                  // global_addr[31:0]
    g0.w = (unsigned)((ga >> 32) & 0x1ffffffu)            // global_addr[56:32]
         | (2u << 30);                                    // type = 2 ("image")
    uint8s g1;
    g1[0] = (1u << 16)                                    // data_size = 2 bytes
          | (1u << 20)                                    // pad_enable
          | (pad_interval_code << 22)
          | (pad_amount_code << 25);
    g1[1] = (tile_d0 & 0xffffu) << 16;                    // tensor_dim0[15:0]
    g1[2] = ((tile_d0 >> 16) & 0xffffu)                   // tensor_dim0[31:16]
          | ((tile_d1 & 0xffffu) << 16);                  // tensor_dim1[15:0]
    g1[3] = ((tile_d1 >> 16) & 0xffffu)                   // tensor_dim1[31:16]
          | ((tile_d0 & 0xffffu) << 16);                  // tile_dim0
    g1[4] = tile_d1 & 0xffffu;                            // tile_dim1 (tile_dim2=0)
    g1[5] = row_stride_elems;                             // tensor_dim0_stride[31:0]
    g1[6] = 0u;                                           // stride[47:32], dim1_stride lo
    g1[7] = 0u;
    asm volatile("tensor_load_to_lds %0, %1" :: "s"(g0), "s"(g1) : "memory");
}

// ---------------------------------------------------------------------------
// GEMM: C[M,N] = A[M,K] * B[K,N].  B always fp32 (weights).
//   A_BF16:  A is bf16 -> tile staged via async global->LDS DMA (no VALU)
//   OUT_BF16: C written as bf16
// Block = 256 threads (8 waves), tile 128x128, K-step 32, waves 2x4,
// each wave computes 64x32 = 4x2 WMMA tiles.
// ---------------------------------------------------------------------------
#define GA_STR 40   // ushorts per A-tile row (32 + 8 pad) = 80B: 16B-aligned rows
#define GB_STR 34   // ushorts per Bt-tile row (32 + 2 pad)

template<bool A_BF16, bool OUT_BF16>
__global__ __launch_bounds__(256) void gemm_wmma(
    const void* __restrict__ Ap, const float* __restrict__ B,
    void* __restrict__ Cp, int M, int N, int K)
{
    __shared__ __attribute__((aligned(16))) unsigned short As[128 * GA_STR];  // [m][k]
    __shared__ __attribute__((aligned(16))) unsigned short Bs[128 * GB_STR];  // [n][k]

    const int tid   = threadIdx.x;
    const int lane  = tid & 31;
    const int wid   = tid >> 5;
    const int waveM = wid >> 2;        // 0..1
    const int waveN = wid & 3;         // 0..3
    const int hi    = lane >> 4;
    const int l15   = lane & 15;
    const int bm    = blockIdx.y * 128;
    const int bn    = blockIdx.x * 128;

    floatx8 acc[4][2];
#pragma unroll
    for (int mt = 0; mt < 4; ++mt)
#pragma unroll
        for (int nt = 0; nt < 2; ++nt)
#pragma unroll
            for (int e = 0; e < 8; ++e) acc[mt][nt][e] = 0.0f;

    const unsigned* As32 = (const unsigned*)As;
    const unsigned* Bs32 = (const unsigned*)Bs;
    const unsigned as_base = lds_off32(&As[0]);
    const int akb = hi ? 4 : 0;   // dword offset in row (hi lanes: K=8..15 first)
    const int bkb = hi ? 4 : 0;   // dword offset (hi lanes: K=8..15 first)

    for (int k0 = 0; k0 < K; k0 += 32) {
        // ---- stage A tile (128 x 32) ----
        if constexpr (A_BF16) {
            const unsigned short* A = (const unsigned short*)Ap;
#pragma unroll
            for (int j = 0; j < 2; ++j) {
                int t   = tid * 2 + j;            // 512 b128 transfers
                int row = t >> 2;                 // 4 per row
                int kc8 = (t & 3) * 8;
                async_copy_b128(A + (size_t)(bm + row) * K + k0 + kc8,
                                as_base + (unsigned)(row * GA_STR + kc8) * 2u);
            }
        } else {
            const float* A = (const float*)Ap;
#pragma unroll
            for (int j = 0; j < 4; ++j) {
                int f4  = tid * 4 + j;            // 1024 float4s
                int row = f4 >> 3;                // 8 per row
                int kc  = (f4 & 7) * 4;
                const float4 a4 = *(const float4*)(A + (size_t)(bm + row) * K + k0 + kc);
                int base = row * GA_STR + kc;
                As[base + 0] = f2bf(a4.x);
                As[base + 1] = f2bf(a4.y);
                As[base + 2] = f2bf(a4.z);
                As[base + 3] = f2bf(a4.w);
            }
        }
        // ---- stage B tile (32 x 128) fp32 -> bf16, transposed to [n][k] ----
#pragma unroll
        for (int j = 0; j < 4; ++j) {
            int f4  = tid * 4 + j;
            int row = f4 >> 5;                    // 32 float4 per k-row
            int cc  = (f4 & 31) * 4;
            const float4 b4 = *(const float4*)(B + (size_t)(k0 + row) * N + bn + cc);
            Bs[(cc + 0) * GB_STR + row] = f2bf(b4.x);
            Bs[(cc + 1) * GB_STR + row] = f2bf(b4.y);
            Bs[(cc + 2) * GB_STR + row] = f2bf(b4.z);
            Bs[(cc + 3) * GB_STR + row] = f2bf(b4.w);
        }
        if (k0 + 32 < K)
            __builtin_prefetch(B + (size_t)(k0 + 32 + (tid >> 5)) * N + bn, 0, 1);
        if constexpr (A_BF16) wait_asynccnt0();
        __syncthreads();

        // ---- fragments (ISA 7.12.2 layouts) ----
        Frag af[4], bfr[2];
#pragma unroll
        for (int mt = 0; mt < 4; ++mt) {
            int m    = waveM * 64 + mt * 16 + l15;
            int base = m * (GA_STR / 2) + akb;
#pragma unroll
            for (int i = 0; i < 4; ++i) {
                af[mt].u[i]     = As32[base + i];      // K = kb .. kb+7
                af[mt].u[4 + i] = As32[base + 8 + i];  // K = kb+16 .. kb+23
            }
        }
#pragma unroll
        for (int nt = 0; nt < 2; ++nt) {
            int n    = waveN * 32 + nt * 16 + l15;
            int base = n * (GB_STR / 2) + bkb;
#pragma unroll
            for (int i = 0; i < 8; ++i) bfr[nt].u[i] = Bs32[base + i + (i & 4)];
        }
#pragma unroll
        for (int mt = 0; mt < 4; ++mt)
#pragma unroll
            for (int nt = 0; nt < 2; ++nt)
                acc[mt][nt] = wmma_bf16(af[mt], bfr[nt], acc[mt][nt]);
        __syncthreads();
    }

    // ---- store C (col = lane&15, row = r + 8*(lane>=16)) ----
#pragma unroll
    for (int mt = 0; mt < 4; ++mt)
#pragma unroll
        for (int nt = 0; nt < 2; ++nt) {
            int n     = bn + waveN * 32 + nt * 16 + l15;
            int mbase = bm + waveM * 64 + mt * 16 + 8 * hi;
#pragma unroll
            for (int r = 0; r < 8; ++r) {
                if constexpr (OUT_BF16)
                    ((unsigned short*)Cp)[(size_t)(mbase + r) * N + n] = f2bf(acc[mt][nt][r]);
                else
                    ((float*)Cp)[(size_t)(mbase + r) * N + n] = acc[mt][nt][r];
            }
        }
}

// ---------------------------------------------------------------------------
// RoPE on bf16 in place; pairs (d, d+64), freq = s * base^(-d/64).
// outscale folds the 1/sqrt(HEAD_DIM) score scale into Q.
// ---------------------------------------------------------------------------
__global__ __launch_bounds__(256) void rope_bf16(unsigned short* __restrict__ x,
                                                 int S, int nheads, float outscale)
{
    int idx   = blockIdx.x * blockDim.x + threadIdx.x;
    int total = S * nheads * 64;
    if (idx >= total) return;
    int i = idx & 63;
    int t = idx >> 6;
    int h = t % nheads;
    int s = t / nheads;
    const float ln_base_over_64 = 0.14391156f;   // ln(10000)/64
    float invf = __expf(-(float)i * ln_base_over_64);
    float fr   = (float)s * invf;
    float c = __cosf(fr), sn = __sinf(fr);
    unsigned short* p = x + (size_t)s * (nheads * 128) + h * 128 + i;
    float x1 = bf2f(p[0]), x2 = bf2f(p[64]);
    p[0]  = f2bf((x1 * c - x2 * sn) * outscale);
    p[64] = f2bf((x2 * c + x1 * sn) * outscale);
}

// ---------------------------------------------------------------------------
// Flash attention (GQA), all-bf16 inputs. Block = 128 threads (4 waves), one
// block per (64-query tile, head); each wave owns 16 query rows; key tile 32.
// Q tile: per-lane async DMA.  K tile: TDM tensor_load_to_lds with hardware
// row padding (256B data + 16B pad = 272B = KS_STR).  V: ds transpose.
// Output (bf16) written in place over Q workspace.
// ---------------------------------------------------------------------------
#define QS_STR 136   // 128 + 8 pad ushorts = 272B rows (16B-aligned)
#define KS_STR 136
#define VS_STR 36    // 32 + 4 pad
#define PS_STR 36

__global__ __launch_bounds__(128) void flash_attn_wmma(
    const unsigned short* __restrict__ Q, const unsigned short* __restrict__ Kv,
    const unsigned short* __restrict__ V, const float* __restrict__ mask,
    unsigned short* __restrict__ O, int S)
{
    __shared__ __attribute__((aligned(16))) unsigned short Qs[64 * QS_STR];   // [q][d]
    __shared__ __attribute__((aligned(16))) unsigned short Ks[32 * KS_STR];   // [key][d]
    __shared__ __attribute__((aligned(16))) unsigned short Vs[128 * VS_STR];  // [d][key]
    __shared__ __attribute__((aligned(16))) unsigned short Ps[64 * PS_STR];   // [q][key]

    const int h   = blockIdx.y;
    const int hk  = h >> 2;                        // 32 heads -> 8 kv heads
    const int qt  = blockIdx.x;
    const int tid = threadIdx.x;
    const int lane = tid & 31, wid = tid >> 5;
    const int hi = lane >> 4, l15 = lane & 15;

    const unsigned qs_base = lds_off32(&Qs[0]);
    const unsigned ks_base = lds_off32(&Ks[0]);

    // ---- async-stage Q tile (64 x 128 bf16): 1024 b128s, 8 per thread ----
#pragma unroll
    for (int j = 0; j < 8; ++j) {
        int t   = j * 128 + tid;
        int row = t >> 4;                          // 16 b128 per row
        int kc8 = (t & 15) * 8;
        async_copy_b128(Q + (size_t)(qt * 64 + row) * 4096 + h * 128 + kc8,
                        qs_base + (unsigned)(row * QS_STR + kc8) * 2u);
    }
    wait_asynccnt0();
    __syncthreads();

    // ---- preload Q A-fragments for the 4 d-steps of 32 ----
    Frag qf[4];
    {
        const unsigned* Qs32 = (const unsigned*)Qs;
        int m   = wid * 16 + l15;
        int akb = hi ? 4 : 0;
#pragma unroll
        for (int kk = 0; kk < 4; ++kk) {
            int base = m * (QS_STR / 2) + kk * 16 + akb;
#pragma unroll
            for (int i = 0; i < 4; ++i) {
                qf[kk].u[i]     = Qs32[base + i];
                qf[kk].u[4 + i] = Qs32[base + 8 + i];
            }
        }
    }

    floatx8 oacc[8];
    float mrun[8], lrun[8];
#pragma unroll
    for (int d = 0; d < 8; ++d)
#pragma unroll
        for (int e = 0; e < 8; ++e) oacc[d][e] = 0.0f;
#pragma unroll
    for (int r = 0; r < 8; ++r) { mrun[r] = -1.0e30f; lrun[r] = 0.0f; }

    const unsigned* Ks32 = (const unsigned*)Ks;
    const unsigned* Vs32 = (const unsigned*)Vs;
    const unsigned* Ps32 = (const unsigned*)Ps;

    const int nTiles = S / 32;
    for (int kt = 0; kt < nTiles; ++kt) {
        __syncthreads();
        // ---- TDM: K tile (32 rows x 128 bf16, row stride 1024 elems) ----
        // pad_interval code 5 = every 64 DWORDs (256B row), pad_amount code 3
        // = 4 DWORDs (16B)  ->  LDS row stride 272B == KS_STR ushorts.
        if (wid == 0) {
            tdm_load_2d_bf16(ks_base,
                             Kv + (size_t)(kt * 32) * 1024 + hk * 128,
                             /*tile_d0=*/128, /*tile_d1=*/32,
                             /*row_stride=*/1024,
                             /*pad_interval=*/5u, /*pad_amount=*/3u);
        }
        // ---- stage V tile (32 x 128 bf16) transposed to [d][key] ----
#pragma unroll
        for (int j = 0; j < 4; ++j) {
            int t   = j * 128 + tid;
            int row = t >> 4;
            int kc8 = (t & 15) * 8;
            const uint4 vv = *(const uint4*)(V + (size_t)(kt * 32 + row) * 1024 + hk * 128 + kc8);
            Vs[(kc8 + 0) * VS_STR + row] = (unsigned short)(vv.x & 0xffffu);
            Vs[(kc8 + 1) * VS_STR + row] = (unsigned short)(vv.x >> 16);
            Vs[(kc8 + 2) * VS_STR + row] = (unsigned short)(vv.y & 0xffffu);
            Vs[(kc8 + 3) * VS_STR + row] = (unsigned short)(vv.y >> 16);
            Vs[(kc8 + 4) * VS_STR + row] = (unsigned short)(vv.z & 0xffffu);
            Vs[(kc8 + 5) * VS_STR + row] = (unsigned short)(vv.z >> 16);
            Vs[(kc8 + 6) * VS_STR + row] = (unsigned short)(vv.w & 0xffffu);
            Vs[(kc8 + 7) * VS_STR + row] = (unsigned short)(vv.w >> 16);
        }
        if (kt + 1 < nTiles)
            __builtin_prefetch(Kv + (size_t)((kt + 1) * 32 + (tid >> 2)) * 1024 + hk * 128, 0, 1);
        if (wid == 0)
            __builtin_amdgcn_s_wait_tensorcnt(0);
        __syncthreads();   // publishes TDM-written K tile + ds-written V tile

        // ---- S = Q K^T (16q x 32key per wave) ----
        floatx8 sacc[2];
#pragma unroll
        for (int nt = 0; nt < 2; ++nt)
#pragma unroll
            for (int e = 0; e < 8; ++e) sacc[nt][e] = 0.0f;

        const int bkb = hi ? 4 : 0;
#pragma unroll
        for (int nt = 0; nt < 2; ++nt) {
            int key = nt * 16 + l15;
#pragma unroll
            for (int kk = 0; kk < 4; ++kk) {
                Frag kf;
                int base = key * (KS_STR / 2) + kk * 16 + bkb;
#pragma unroll
                for (int i = 0; i < 8; ++i) kf.u[i] = Ks32[base + i + (i & 4)];
                sacc[nt] = wmma_bf16(qf[kk], kf, sacc[nt]);
            }
        }

        // ---- additive mask (per key column) ----
#pragma unroll
        for (int nt = 0; nt < 2; ++nt) {
            float mk = mask[kt * 32 + nt * 16 + l15];
#pragma unroll
            for (int r = 0; r < 8; ++r) sacc[nt][r] += mk;
        }

        // ---- online softmax (rows live in 16-lane halves) ----
        float corr[8];
#pragma unroll
        for (int r = 0; r < 8; ++r) {
            float v = fmaxf(sacc[0][r], sacc[1][r]);
#pragma unroll
            for (int off = 1; off < 16; off <<= 1)
                v = fmaxf(v, __shfl_xor(v, off, 32));
            float mnew = fmaxf(mrun[r], v);
            corr[r] = __expf(mrun[r] - mnew);
            mrun[r] = mnew;
        }
#pragma unroll
        for (int r = 0; r < 8; ++r) {
            float p0 = __expf(sacc[0][r] - mrun[r]);
            float p1 = __expf(sacc[1][r] - mrun[r]);
            sacc[0][r] = p0; sacc[1][r] = p1;
            float v = p0 + p1;
#pragma unroll
            for (int off = 1; off < 16; off <<= 1)
                v += __shfl_xor(v, off, 32);
            lrun[r] = lrun[r] * corr[r] + v;
        }
#pragma unroll
        for (int d = 0; d < 8; ++d)
#pragma unroll
            for (int r = 0; r < 8; ++r) oacc[d][r] *= corr[r];

        // ---- P -> LDS (C-layout to row-major bf16) ----
#pragma unroll
        for (int nt = 0; nt < 2; ++nt)
#pragma unroll
            for (int r = 0; r < 8; ++r)
                Ps[(wid * 16 + r + 8 * hi) * PS_STR + nt * 16 + l15] = f2bf(sacc[nt][r]);
        __syncthreads();

        // ---- O += P V ----
        Frag pf;
        {
            int m    = wid * 16 + l15;
            int akb  = hi ? 4 : 0;
            int base = m * (PS_STR / 2) + akb;
#pragma unroll
            for (int i = 0; i < 4; ++i) {
                pf.u[i]     = Ps32[base + i];
                pf.u[4 + i] = Ps32[base + 8 + i];
            }
        }
        const int vkb = hi ? 4 : 0;
#pragma unroll
        for (int d = 0; d < 8; ++d) {
            Frag vf;
            int n    = d * 16 + l15;
            int base = n * (VS_STR / 2) + vkb;
#pragma unroll
            for (int i = 0; i < 8; ++i) vf.u[i] = Vs32[base + i + (i & 4)];
            oacc[d] = wmma_bf16(pf, vf, oacc[d]);
        }
    }

    // ---- normalize + store bf16 ----
#pragma unroll
    for (int d = 0; d < 8; ++d) {
        int col  = h * 128 + d * 16 + l15;
        int mrow = qt * 64 + wid * 16 + 8 * hi;
#pragma unroll
        for (int r = 0; r < 8; ++r)
            O[(size_t)(mrow + r) * 4096 + col] = f2bf(oacc[d][r] / lrun[r]);
    }
}

// ---------------------------------------------------------------------------
// Host-side orchestration (graph-capture safe: only kernel launches on stream)
// ---------------------------------------------------------------------------
extern "C" void kernel_launch(void* const* d_in, const int* in_sizes, int n_in,
                              void* d_out, int out_size, void* d_ws, size_t ws_size,
                              hipStream_t stream)
{
    const int S = 2048, H = 4096, QN = 4096, KVN = 1024;
    const float* hidden = (const float*)d_in[0];
    const float* mask   = (const float*)d_in[1];
    const float* Wq     = (const float*)d_in[2];
    const float* Wk     = (const float*)d_in[3];
    const float* Wv     = (const float*)d_in[4];
    const float* Wo     = (const float*)d_in[5];
    float* out          = (float*)d_out;

    unsigned short* q_ws = (unsigned short*)d_ws;       // [S, 4096] bf16
    unsigned short* k_ws = q_ws + (size_t)S * QN;       // [S, 1024] bf16
    unsigned short* v_ws = k_ws + (size_t)S * KVN;      // [S, 1024] bf16

    // QKV projections (fp32 activations/weights -> bf16 outputs)
    gemm_wmma<false, true><<<dim3(QN / 128,  S / 128), 256, 0, stream>>>(hidden, Wq, q_ws, S, QN,  H);
    gemm_wmma<false, true><<<dim3(KVN / 128, S / 128), 256, 0, stream>>>(hidden, Wk, k_ws, S, KVN, H);
    gemm_wmma<false, true><<<dim3(KVN / 128, S / 128), 256, 0, stream>>>(hidden, Wv, v_ws, S, KVN, H);

    // RoPE on Q (scale folded in) and K
    rope_bf16<<<(S * 32 * 64 + 255) / 256, 256, 0, stream>>>(q_ws, S, 32, 0.08838834764831845f);
    rope_bf16<<<(S * 8  * 64 + 255) / 256, 256, 0, stream>>>(k_ws, S, 8, 1.0f);

    // Flash attention (bf16 in/out, output in place over Q workspace)
    flash_attn_wmma<<<dim3(S / 64, 32), 128, 0, stream>>>(q_ws, k_ws, v_ws, mask, q_ws, S);

    // Output projection (bf16 A staged via async DMA, fp32 result)
    gemm_wmma<true, false><<<dim3(H / 128, S / 128), 256, 0, stream>>>(q_ws, Wo, out, S, H, H);
}
